// GraphAttentionLayer_27616639713548
// MI455X (gfx1250) — matrix operand order repaired and checked by reading, not compiled, not run on hardware
//
#include <hip/hip_runtime.h>
#include <hip/hip_bf16.h>

// ---- problem constants (from setup_inputs) ----
#define BB   8
#define FIN  64
#define TT   12
#define NN   512
#define FOUT 64
#define NEG_INF -9000000000000000.0f

// ---- WMMA vector types (gfx1250, wave32) ----
typedef __attribute__((ext_vector_type(16))) __bf16 v16bf;
typedef __attribute__((ext_vector_type(8)))  __bf16 bf16x8;
typedef __attribute__((ext_vector_type(8)))  float  v8f;

union Frag16 { v16bf v; bf16x8 h[2]; };

// ---- LDS layout (bytes). Row strides chosen so every 8-half chunk is 16B aligned.
#define SWT_STRIDE  72            // halves per row of W^T   (64 + pad)
#define SX_STRIDE   72            // halves per row of x     (64 + pad)
#define SWH_STRIDE  520           // halves per row of Wh^T  (512 + pad), 1040B % 16 == 0
#define ET_STRIDE   520           // halves per row of prob tile

#define OFF_SWT   0                                   // 64  * 72  * 2 = 9216
#define OFF_SX    (OFF_SWT  + 64 * SWT_STRIDE * 2)    // 512 * 72  * 2 = 73728
#define OFF_SWH   (OFF_SX   + NN * SX_STRIDE  * 2)    // 64  * 520 * 2 = 66560
#define OFF_SF1   (OFF_SWH  + 64 * SWH_STRIDE * 2)    // 512 * 4       = 2048
#define OFF_SF2   (OFF_SF1  + NN * 4)
#define OFF_SA    (OFF_SF2  + NN * 4)                 // 128 * 4       = 512
#define OFF_RINV  (OFF_SA   + 128 * 4)                // 8*16*4        = 512
#define OFF_ET    (OFF_RINV + 8 * 16 * 4)             // 8 * 16 * 520 * 2 = 133120
#define SMEM_TOTAL (OFF_ET + 8 * 16 * ET_STRIDE * 2)  // = 287744 bytes (281 KB <= 320 KB)

__device__ __forceinline__ v8f wmma_bf16(const v16bf& a, const v16bf& b, const v8f& c) {
    return __builtin_amdgcn_wmma_f32_16x16x32_bf16(
        /*neg_a=*/false, a, /*neg_b=*/false, b,
        /*c_mod=*/(short)0, c, /*reuse_a=*/false, /*reuse_b=*/false);
}

__global__ __launch_bounds__(256)
void gat_kernel(const float* __restrict__ input,   // (B, F_in, T, N)
                const int*   __restrict__ adj,     // (B, T, N, N)
                const float* __restrict__ Wmat,    // (F_in, F_out)
                const float* __restrict__ avec,    // (2*F_out, 1)
                float*       __restrict__ out)     // (B, T, N, F_out)
{
    extern __shared__ __attribute__((aligned(16))) char smem[];
    __bf16* sWT  = (__bf16*)(smem + OFF_SWT);   // [FOUT][SWT_STRIDE]  W^T, bf16
    __bf16* sX   = (__bf16*)(smem + OFF_SX);    // [NN][SX_STRIDE]     x,   bf16
    __bf16* sWhT = (__bf16*)(smem + OFF_SWH);   // [FOUT][SWH_STRIDE]  Wh^T bf16
    float*  sF1  = (float*)(smem + OFF_SF1);    // [NN]
    float*  sF2  = (float*)(smem + OFF_SF2);    // [NN]
    float*  sA   = (float*)(smem + OFF_SA);     // [128] attention vec
    float*  sRinv= (float*)(smem + OFF_RINV);   // [8][16] 1/rowsum per wave tile
    __bf16* sET  = (__bf16*)(smem + OFF_ET);    // [8][16][ET_STRIDE] unnormalized probs

    const int tid  = threadIdx.x;
    const int lane = tid & 31;
    const int wave = tid >> 5;
    const int bi   = blockIdx.x / TT;
    const int ti   = blockIdx.x % TT;

    // ---------------- Stage 0: stage x, W^T, a into LDS ----------------
    {
        // x: input[b][f][t][n] -> sX[n][f] (bf16). 4 threads per f, 128 n each.
        const int f = tid >> 2;
        const int q = tid & 3;
        const float* src = input + (((size_t)bi * FIN + f) * TT + ti) * NN + q * 128;
        #pragma unroll 4
        for (int n4 = 0; n4 < 32; ++n4) {
            float4 v = *(const float4*)(src + n4 * 4);
            int n = q * 128 + n4 * 4;
            sX[(n + 0) * SX_STRIDE + f] = (__bf16)v.x;
            sX[(n + 1) * SX_STRIDE + f] = (__bf16)v.y;
            sX[(n + 2) * SX_STRIDE + f] = (__bf16)v.z;
            sX[(n + 3) * SX_STRIDE + f] = (__bf16)v.w;
        }
        // W: (F_in,F_out) row-major -> sWT[f_out][f_in]
        const int k = tid >> 2;
        const float* wsrc = Wmat + k * FOUT + q * 16;
        #pragma unroll
        for (int n4 = 0; n4 < 4; ++n4) {
            float4 v = *(const float4*)(wsrc + n4 * 4);
            int n = q * 16 + n4 * 4;
            sWT[(n + 0) * SWT_STRIDE + k] = (__bf16)v.x;
            sWT[(n + 1) * SWT_STRIDE + k] = (__bf16)v.y;
            sWT[(n + 2) * SWT_STRIDE + k] = (__bf16)v.z;
            sWT[(n + 3) * SWT_STRIDE + k] = (__bf16)v.w;
        }
        if (tid < 128) sA[tid] = avec[tid];
    }
    __syncthreads();

    // ---------------- Stage A: Wh = x @ W  (512x64 = 32 Mtiles x 4 Ntiles) ----
    // wave32 WMMA fragment indices (ISA 7.12.2):
    //   A 16x32 bf16: lane m=lane&15, hi=lane>>4; halves 0..7 -> K=k0+8*hi+h,
    //                 halves 8..15 -> K=k0+16+8*hi+(h-8)   (contiguous b128 pairs)
    //   B 32x16 bf16: lane n=lane&15, hi=lane>>4; same K mapping as A
    //   D 16x16 f32:  vgpr r, (M=r+8*hi, N=lane&15)
    const int m  = lane & 15;
    const int hi = lane >> 4;
    #pragma unroll
    for (int g = 0; g < 4; ++g) {
        const int mtile = wave + g * 8;       // 0..31
        const int node0 = mtile * 16;
        const __bf16* arow = sX + (size_t)(node0 + m) * SX_STRIDE;
        #pragma unroll
        for (int nt = 0; nt < 4; ++nt) {
            v8f acc = {};
            const __bf16* brow = sWT + (size_t)(nt * 16 + m) * SWT_STRIDE;
            #pragma unroll
            for (int kc = 0; kc < 2; ++kc) {
                const int k0 = kc * 32;
                Frag16 A, Bf;
                A.h[0]  = *(const bf16x8*)(arow + k0 + 8 * hi);
                A.h[1]  = *(const bf16x8*)(arow + k0 + 16 + 8 * hi);
                Bf.h[0] = *(const bf16x8*)(brow + k0 + 8 * hi);
                Bf.h[1] = *(const bf16x8*)(brow + k0 + 16 + 8 * hi);
                acc = wmma_bf16(A.v, Bf.v, acc);
            }
            #pragma unroll
            for (int r = 0; r < 8; ++r) {
                const int mm = r + 8 * hi;
                const int f  = nt * 16 + m;
                sWhT[(size_t)f * SWH_STRIDE + node0 + mm] = (__bf16)acc[r];
            }
        }
    }
    __syncthreads();

    // ---------------- Stage F: f1/f2 = Wh . a1 / a2 ----------------
    #pragma unroll
    for (int nn = 0; nn < 2; ++nn) {
        const int node = tid * 2 + nn;
        float s1 = 0.f, s2 = 0.f;
        #pragma unroll 8
        for (int f = 0; f < FOUT; ++f) {
            float wh = (float)sWhT[(size_t)f * SWH_STRIDE + node];
            s1 += wh * sA[f];
            s2 += wh * sA[FOUT + f];
        }
        sF1[node] = s1;
        sF2[node] = s2;
    }
    __syncthreads();

    // ---------------- Stage B: attn softmax + attn @ Wh + ELU ----------------
    __bf16* eT = sET + (size_t)wave * 16 * ET_STRIDE;   // wave-private 16x512 tile
    const size_t adjBT = ((size_t)bi * TT + ti) * NN;   // row-block base in adj
    const size_t outBT = ((size_t)bi * TT + ti) * NN;   // row base in out

    #pragma unroll 1
    for (int g = 0; g < 4; ++g) {
        const int mtile = wave + g * 8;
        const int row0  = mtile * 16;

        // --- masked logits + online row max/sum (wave32 shuffles), adj read ONCE ---
        #pragma unroll 1
        for (int r = 0; r < 16; ++r) {
            const int i = row0 + r;
            const size_t rb = (adjBT + i) * NN;
            __builtin_prefetch(adj + rb + NN, 0, 0);   // global_prefetch_b8: next row
            const float f1i = sF1[i];
            float mx = -3.0e38f;
            float ev[16];
            #pragma unroll
            for (int c = 0; c < 16; ++c) {
                const int col = c * 32 + lane;
                const int av  = adj[rb + col];
                float e = f1i + sF2[col];
                e = (e > 0.f) ? e : 0.2f * e;          // leaky_relu(0.2)
                if (av <= 0) e = NEG_INF;
                ev[c] = e;
                mx = fmaxf(mx, e);
            }
            #pragma unroll
            for (int off = 16; off > 0; off >>= 1)
                mx = fmaxf(mx, __shfl_xor(mx, off, 32));
            float s = 0.f;
            #pragma unroll
            for (int c = 0; c < 16; ++c) {
                float p = __expf(ev[c] - mx);
                s += p;
                eT[(size_t)r * ET_STRIDE + c * 32 + lane] = (__bf16)p;
            }
            #pragma unroll
            for (int off = 16; off > 0; off >>= 1)
                s += __shfl_xor(s, off, 32);
            if (lane == 0) sRinv[wave * 16 + r] = 1.f / s;
        }

        // --- h tile = P(16x512) @ Wh(512x64), 4 N-tiles, K in 16 steps of 32 ---
        #pragma unroll
        for (int nt = 0; nt < 4; ++nt) {
            v8f acc = {};
            const __bf16* brow = sWhT + (size_t)(nt * 16 + m) * SWH_STRIDE;
            const __bf16* arow = eT + (size_t)m * ET_STRIDE;
            #pragma unroll 4
            for (int kc = 0; kc < 16; ++kc) {
                const int k0 = kc * 32;
                Frag16 A, Bf;
                A.h[0]  = *(const bf16x8*)(arow + k0 + 8 * hi);
                A.h[1]  = *(const bf16x8*)(arow + k0 + 16 + 8 * hi);
                Bf.h[0] = *(const bf16x8*)(brow + k0 + 8 * hi);
                Bf.h[1] = *(const bf16x8*)(brow + k0 + 16 + 8 * hi);
                acc = wmma_bf16(A.v, Bf.v, acc);
            }
            #pragma unroll
            for (int r = 0; r < 8; ++r) {
                const int mm = r + 8 * hi;
                const float rv = sRinv[wave * 16 + mm];
                float v = acc[r] * rv;
                v = (v > 0.f) ? v : (__expf(v) - 1.0f);   // elu, alpha=1
                out[(outBT + row0 + mm) * FOUT + nt * 16 + m] = v;
            }
        }
    }
}

extern "C" void kernel_launch(void* const* d_in, const int* in_sizes, int n_in,
                              void* d_out, int out_size, void* d_ws, size_t ws_size,
                              hipStream_t stream) {
    (void)in_sizes; (void)n_in; (void)out_size; (void)d_ws; (void)ws_size;
    const float* input = (const float*)d_in[0];
    const int*   adj   = (const int*)d_in[1];
    const float* Wmat  = (const float*)d_in[2];
    const float* avec  = (const float*)d_in[3];
    float*       out   = (float*)d_out;

    dim3 grid(BB * TT);   // 96 workgroups, one per (b,t)
    dim3 block(256);      // 8 waves (wave32)
    gat_kernel<<<grid, block, SMEM_TOTAL, stream>>>(input, adj, Wmat, avec, out);
}
